// Attention_18554258718870
// MI455X (gfx1250) — matrix operand order repaired and checked by reading, not compiled, not run on hardware
//
#include <hip/hip_runtime.h>
#include <math.h>

// ---------------- constants ----------------
#define BATCH   64
#define NTOK    260
#define NPAD    288          // padded token count: 18 tiles of 16, 9 K-chunks of 32
#define DIMC    768
#define NHEAD   12
#define HDIM    64
#define BN      (BATCH * NTOK)      // 16640, divisible by 16
#define MTILES  (BN / 16)           // 1040
#define MGROUPS (MTILES / 4)        // 260 (4 waves x 16 rows = 64 M rows per block)
#define TABLE_T 961

typedef __attribute__((ext_vector_type(16))) _Float16 v16h;
typedef __attribute__((ext_vector_type(8)))  float    v8f;
typedef __attribute__((ext_vector_type(4)))  unsigned int v4u;
typedef __attribute__((ext_vector_type(8)))  int      v8i;
typedef __attribute__((ext_vector_type(4)))  int      v4i;

union ABFrag { v16h h; uint4 u[2]; };

#define WMMA(Af, Bf, Cacc) \
  __builtin_amdgcn_wmma_f32_16x16x32_f16(false, (Af), false, (Bf), (short)0, (Cacc), false, false)

#if __has_builtin(__builtin_amdgcn_tensor_load_to_lds)
#define HAVE_TDM 1
#else
#define HAVE_TDM 0
#endif

// LDS byte offset of a generic pointer to __shared__ (aperture: LDS offset = addr[31:0])
__device__ __forceinline__ unsigned lds_off(const void* p) {
  return (unsigned)(unsigned long long)(uintptr_t)p;
}

#if HAVE_TDM
// Issue a TDM 2D tile load: tile_d0 elements/row (2B each), tile_d1 rows,
// row stride `stride0` elements. tensor_dim == tile_dim so OOB logic is inert.
__device__ __forceinline__ void tdm_load_2d(unsigned ldst, const void* gsrc,
                                            unsigned tile_d0, unsigned tile_d1,
                                            unsigned stride0) {
  unsigned long long ga = (unsigned long long)(uintptr_t)gsrc;
  v4u g0;
  g0[0] = 1u;                                            // count=1, user descriptor
  g0[1] = ldst;                                          // lds_addr (bytes)
  g0[2] = (unsigned)(ga & 0xFFFFFFFFu);                  // global_addr[31:0]
  g0[3] = (unsigned)((ga >> 32) & 0x1FFFFFFu) | (2u << 30); // addr[56:32] | type=2
  v8i g1;
  g1[0] = (int)(1u << 16);                               // wg_mask=0, data_size=1 (2B)
  g1[1] = (int)((tile_d0 & 0xFFFFu) << 16);              // tensor_dim0[15:0] @ bit48
  g1[2] = (int)(((tile_d0 >> 16) & 0xFFFFu) | ((tile_d1 & 0xFFFFu) << 16)); // td0 hi | td1 lo
  g1[3] = (int)(((tile_d1 >> 16) & 0xFFFFu) | ((tile_d0 & 0xFFFFu) << 16)); // td1 hi | tile_dim0
  g1[4] = (int)(tile_d1 & 0xFFFFu);                      // tile_dim1, tile_dim2=0
  g1[5] = (int)stride0;                                  // tensor_dim0_stride[31:0]
  g1[6] = 0;                                             // stride0[47:32], stride1 lo
  g1[7] = 0;
  v4i z4 = {0, 0, 0, 0};
#if __clang_major__ >= 23
  v8i z8 = {0, 0, 0, 0, 0, 0, 0, 0};
  __builtin_amdgcn_tensor_load_to_lds(g0, g1, z4, z4, z8, 0);
#else
  __builtin_amdgcn_tensor_load_to_lds(g0, g1, z4, z4, 0);
#endif
}
#endif

// ---------------- elementwise helpers ----------------
__global__ void cvt_f32_to_f16(const float* __restrict__ in, _Float16* __restrict__ out, int n) {
  int i = blockIdx.x * blockDim.x + threadIdx.x;
  if (i < n) out[i] = (_Float16)in[i];
}

__global__ void zero_u4(uint4* __restrict__ p, int n4) {
  int i = blockIdx.x * blockDim.x + threadIdx.x;
  if (i < n4) p[i] = make_uint4(0u, 0u, 0u, 0u);
}

// ---------------- position-bias MLP: bt[961,12] ----------------
__global__ void bias_mlp(const float* __restrict__ rel, const float* __restrict__ w1,
                         const float* __restrict__ b1, const float* __restrict__ w2,
                         float* __restrict__ bt) {
  int t = blockIdx.x * blockDim.x + threadIdx.x;
  if (t >= TABLE_T) return;
  float ry = rel[2 * t], rx = rel[2 * t + 1];
  float g[32];
#pragma unroll
  for (int j = 0; j < 32; ++j) {
    float hv = ry * w1[2 * j] + rx * w1[2 * j + 1] + b1[j];
    g[j] = 0.5f * hv * (1.0f + erff(hv * 0.70710678118654752f));   // exact GELU
  }
#pragma unroll
  for (int hh = 0; hh < NHEAD; ++hh) {
    float s = 0.f;
#pragma unroll
    for (int j = 0; j < 32; ++j) s += g[j] * w2[hh * 32 + j];
    bt[t * NHEAD + hh] = s;
  }
}

// ============ block-cooperative WMMA GEMM core ============
// Block = 4 waves; block computes 64(M) x 64(N). B tile (64 rows x 32 K-elems)
// is staged in LDS (double-buffered) via TDM by wave 0; A frags are per-wave
// register loads pipelined one K-chunk ahead.
template <typename StoreFn>
__device__ __forceinline__ void gemm64x64(const _Float16* __restrict__ Amat,
                                          const _Float16* __restrict__ Wmat,
                                          int mtile, int n0, StoreFn&& store) {
  __shared__ __align__(16) _Float16 sB[2][64 * 32];   // 2 x 4KB
  const int lane = threadIdx.x & 31;
  const int wave = threadIdx.x >> 5;
  const int ln   = lane & 15;
  const int half = lane >> 4;
  const _Float16* Arow = Amat + (size_t)(mtile * 16 + ln) * DIMC;
  const _Float16* Wrow = Wmat + (size_t)n0 * DIMC;

#if HAVE_TDM
  if (wave == 0) tdm_load_2d(lds_off(&sB[0][0]), Wrow, 32, 64, DIMC);
#else
  for (int i = threadIdx.x; i < 256; i += 128) {
    int r = i >> 2, c = i & 3;
    ((uint4*)&sB[0][0])[i] = *(const uint4*)(Wrow + (size_t)r * DIMC + c * 8);
  }
#endif

  v8f acc[4] = {};
  ABFrag a;
  a.u[0] = *(const uint4*)(Arow + half * 16);
  a.u[1] = *(const uint4*)(Arow + half * 16 + 8);

  for (int kc = 0; kc < 24; ++kc) {
#if HAVE_TDM
    if (wave == 0) __builtin_amdgcn_s_wait_tensorcnt(0);
#endif
    __syncthreads();                       // sB[kc&1] ready for everyone
    if (kc + 1 < 24) {
#if HAVE_TDM
      if (wave == 0)
        tdm_load_2d(lds_off(&sB[(kc + 1) & 1][0]), Wrow + (kc + 1) * 32, 32, 64, DIMC);
#else
      for (int i = threadIdx.x; i < 256; i += 128) {
        int r = i >> 2, c = i & 3;
        ((uint4*)&sB[(kc + 1) & 1][0])[i] =
            *(const uint4*)(Wrow + (size_t)r * DIMC + (kc + 1) * 32 + c * 8);
      }
#endif
    }
    // pipeline next A fragment
    ABFrag an;
    if (kc + 1 < 24) {
      int kbA = (kc + 1) * 32 + half * 16;
      an.u[0] = *(const uint4*)(Arow + kbA);
      an.u[1] = *(const uint4*)(Arow + kbA + 8);
      __builtin_prefetch(Arow + kbA + 32, 0, 1);
    }
    const _Float16* bb = &sB[kc & 1][0];
#pragma unroll
    for (int j = 0; j < 4; ++j) {
      ABFrag b;
      const _Float16* br = bb + (size_t)(j * 16 + ln) * 32 + half * 16;
      b.u[0] = *(const uint4*)(br);
      b.u[1] = *(const uint4*)(br + 8);
      acc[j] = WMMA(a.h, b.h, acc[j]);
    }
    a = an;
    __syncthreads();                       // reads done before buffer reuse
  }
  store(acc, mtile, n0, ln, half);
}

// ---------------- QKV GEMM -> Q,K f16 [B,H,NPAD,D], V^T f16 [B,H,D,NPAD] ----------------
__global__ __launch_bounds__(128) void qkv_gemm(
    const _Float16* __restrict__ X, const _Float16* __restrict__ W,
    _Float16* __restrict__ Q, _Float16* __restrict__ K2, _Float16* __restrict__ VT) {
  const int wave = threadIdx.x >> 5;
  int mg = blockIdx.x / 36;
  int n0 = (blockIdx.x % 36) * 64;
  int mtile = mg * 4 + wave;

  gemm64x64(X, W, mtile, n0,
    [&](v8f (&acc)[4], int mt, int nb, int ln, int half) {
#pragma unroll
      for (int j = 0; j < 4; ++j) {
        int o = nb + j * 16 + ln;
        int t = o / DIMC, rem = o % DIMC, h = rem / HDIM, d = rem % HDIM;
#pragma unroll
        for (int g = 0; g < 8; ++g) {
          int m  = mt * 16 + g + half * 8;
          int bb = m / NTOK, nn = m % NTOK;
          _Float16 val = (_Float16)acc[j][g];
          size_t bh = (size_t)bb * NHEAD + h;
          if (t == 0)      Q [(bh * NPAD + nn) * HDIM + d] = val;
          else if (t == 1) K2[(bh * NPAD + nn) * HDIM + d] = val;
          else             VT[(bh * HDIM + d) * NPAD + nn] = val;
        }
      }
    });
}

// ---------------- projection GEMM + bias -> fp32 out ----------------
__global__ __launch_bounds__(128) void proj_gemm(
    const _Float16* __restrict__ A, const _Float16* __restrict__ W,
    const float* __restrict__ bias, float* __restrict__ out) {
  const int wave = threadIdx.x >> 5;
  int mg = blockIdx.x / 12;
  int n0 = (blockIdx.x % 12) * 64;
  int mtile = mg * 4 + wave;

  gemm64x64(A, W, mtile, n0,
    [&](v8f (&acc)[4], int mt, int nb, int ln, int half) {
#pragma unroll
      for (int j = 0; j < 4; ++j) {
        int o = nb + j * 16 + ln;
        float bv = bias[o];
#pragma unroll
        for (int g = 0; g < 8; ++g) {
          int m = mt * 16 + g + half * 8;
          out[(size_t)m * DIMC + o] = acc[j][g] + bv;
        }
      }
    });
}

// ---------------- attention: one block per (b,h), 4 waves ----------------
__global__ __launch_bounds__(128) void attn_kernel(
    const _Float16* __restrict__ Q, const _Float16* __restrict__ K2,
    const _Float16* __restrict__ VT, const float* __restrict__ bt,
    const int* __restrict__ idxt, _Float16* __restrict__ AO) {
  __shared__ __align__(16) _Float16 sP[4][16][NPAD];
  const int lane = threadIdx.x & 31;
  const int wave = threadIdx.x >> 5;
  const int ln   = lane & 15;
  const int half = lane >> 4;
  const int bh = blockIdx.x;
  const int h  = bh % NHEAD;
  const int bi = bh / NHEAD;
  const _Float16* qb = Q  + (size_t)bh * NPAD * HDIM;
  const _Float16* kb = K2 + (size_t)bh * NPAD * HDIM;
  const _Float16* vb = VT + (size_t)bh * HDIM * NPAD;

  for (int qt = wave; qt < 17; qt += 4) {           // 17 query tiles cover 260 rows
    ABFrag a0, a1;
    {
      const _Float16* qr = qb + (size_t)(qt * 16 + ln) * HDIM;
      a0.u[0] = *(const uint4*)(qr + half * 16);
      a0.u[1] = *(const uint4*)(qr + half * 16 + 8);
      a1.u[0] = *(const uint4*)(qr + 32 + half * 16);
      a1.u[1] = *(const uint4*)(qr + 32 + half * 16 + 8);
    }
    // pass 1: S = QK^T * scale + bias -> LDS (f16), pad keys = -inf
    for (int kt = 0; kt < 18; ++kt) {
      const _Float16* kr = kb + (size_t)(kt * 16 + ln) * HDIM;
      ABFrag b0, b1;
      b0.u[0] = *(const uint4*)(kr + half * 16);
      b0.u[1] = *(const uint4*)(kr + half * 16 + 8);
      b1.u[0] = *(const uint4*)(kr + 32 + half * 16);
      b1.u[1] = *(const uint4*)(kr + 32 + half * 16 + 8);
      v8f s = {};
      s = WMMA(a0.h, b0.h, s);
      s = WMMA(a1.h, b1.h, s);
      int kcol = kt * 16 + ln;
#pragma unroll
      for (int g = 0; g < 8; ++g) {
        int qrow = qt * 16 + g + half * 8;
        float val = s[g] * 0.125f;                   // 1/sqrt(64)
        if (kcol >= NTOK) val = -INFINITY;
        else if (qrow < NTOK && qrow >= 4 && kcol >= 4)
          val += bt[idxt[qrow * NTOK + kcol] * NHEAD + h];
        sP[wave][g + half * 8][kcol] = (_Float16)val;
      }
    }
    // pass 2a: row max (cross-lane reduce within each half-wave of 16)
    float m[8];
#pragma unroll
    for (int g = 0; g < 8; ++g) m[g] = -INFINITY;
    for (int kt = 0; kt < 18; ++kt)
#pragma unroll
      for (int g = 0; g < 8; ++g)
        m[g] = fmaxf(m[g], (float)sP[wave][g + half * 8][kt * 16 + ln]);
#pragma unroll
    for (int off = 8; off >= 1; off >>= 1)
#pragma unroll
      for (int g = 0; g < 8; ++g)
        m[g] = fmaxf(m[g], __shfl_xor(m[g], off, 32));
    // pass 2b: P = exp(S - m), row sums, write P back to LDS (f16)
    float sum[8] = {};
    for (int kt = 0; kt < 18; ++kt)
#pragma unroll
      for (int g = 0; g < 8; ++g) {
        float sv = (float)sP[wave][g + half * 8][kt * 16 + ln];
        float e  = (m[g] == -INFINITY) ? 0.f : __expf(sv - m[g]);
        sum[g] += e;
        sP[wave][g + half * 8][kt * 16 + ln] = (_Float16)e;
      }
#pragma unroll
    for (int off = 8; off >= 1; off >>= 1)
#pragma unroll
      for (int g = 0; g < 8; ++g)
        sum[g] += __shfl_xor(sum[g], off, 32);
    float inv[8];
#pragma unroll
    for (int g = 0; g < 8; ++g) inv[g] = 1.0f / sum[g];

    // PV: P[16 x 288] x V^T-tiles -> O[16 x 64]
    v8f o[4] = {};
    for (int kc = 0; kc < 9; ++kc) {
      ABFrag p;
      const _Float16* pr = &sP[wave][ln][0];
      int kbp = kc * 32 + half * 16;
      p.u[0] = *(const uint4*)(pr + kbp);
      p.u[1] = *(const uint4*)(pr + kbp + 8);
#pragma unroll
      for (int nt = 0; nt < 4; ++nt) {
        const _Float16* vr = vb + (size_t)(nt * 16 + ln) * NPAD + kc * 32 + half * 16;
        ABFrag b;
        b.u[0] = *(const uint4*)(vr);
        b.u[1] = *(const uint4*)(vr + 8);
        o[nt] = WMMA(p.h, b.h, o[nt]);
      }
    }
    // store [B, N, H*D] f16 for the projection GEMM
#pragma unroll
    for (int nt = 0; nt < 4; ++nt)
#pragma unroll
      for (int g = 0; g < 8; ++g) {
        int n = qt * 16 + g + half * 8;
        if (n < NTOK)
          AO[((size_t)(bi * NTOK + n)) * DIMC + h * HDIM + nt * 16 + ln] =
              (_Float16)(o[nt][g] * inv[g]);
      }
  }
}

// ---------------- host launch ----------------
extern "C" void kernel_launch(void* const* d_in, const int* in_sizes, int n_in,
                              void* d_out, int out_size, void* d_ws, size_t ws_size,
                              hipStream_t stream) {
  const float* x      = (const float*)d_in[0];
  const float* qkv_w  = (const float*)d_in[1];
  const float* proj_w = (const float*)d_in[2];
  const float* proj_b = (const float*)d_in[3];
  const float* mlp_w1 = (const float*)d_in[4];
  const float* mlp_b1 = (const float*)d_in[5];
  const float* mlp_w2 = (const float*)d_in[6];
  const float* rel    = (const float*)d_in[7];
  const int*   idxt   = (const int*)d_in[8];
  (void)in_sizes; (void)n_in; (void)out_size; (void)ws_size;

  char* ws = (char*)d_ws;
  size_t off = 0;
  _Float16* xh     = (_Float16*)(ws + off); off += (size_t)BN * DIMC * 2;
  _Float16* wqkvh  = (_Float16*)(ws + off); off += (size_t)3 * DIMC * DIMC * 2;
  _Float16* wprojh = (_Float16*)(ws + off); off += (size_t)DIMC * DIMC * 2;
  _Float16* qh     = (_Float16*)(ws + off); off += (size_t)BATCH * NHEAD * NPAD * HDIM * 2;
  _Float16* kh     = (_Float16*)(ws + off); off += (size_t)BATCH * NHEAD * NPAD * HDIM * 2;
  _Float16* vth    = (_Float16*)(ws + off); off += (size_t)BATCH * NHEAD * HDIM * NPAD * 2;
  _Float16* aoh    = (_Float16*)(ws + off); off += (size_t)BN * DIMC * 2;
  float*    btb    = (float*)(ws + off);    off += (size_t)TABLE_T * NHEAD * 4;

  {
    int n = BN * DIMC;
    cvt_f32_to_f16<<<(n + 255) / 256, 256, 0, stream>>>(x, xh, n);
    n = 3 * DIMC * DIMC;
    cvt_f32_to_f16<<<(n + 255) / 256, 256, 0, stream>>>(qkv_w, wqkvh, n);
    n = DIMC * DIMC;
    cvt_f32_to_f16<<<(n + 255) / 256, 256, 0, stream>>>(proj_w, wprojh, n);
  }
  {
    size_t halves = (size_t)3 * BATCH * NHEAD * NPAD * HDIM;
    int n4 = (int)(halves / 8);
    zero_u4<<<(n4 + 255) / 256, 256, 0, stream>>>((uint4*)qh, n4);
  }
  bias_mlp<<<(TABLE_T + 127) / 128, 128, 0, stream>>>(rel, mlp_w1, mlp_b1, mlp_w2, btb);
  qkv_gemm<<<MGROUPS * 36, 128, 0, stream>>>(xh, wqkvh, qh, kh, vth);
  attn_kernel<<<BATCH * NHEAD, 128, 0, stream>>>(qh, kh, vth, btb, idxt, aoh);
  proj_gemm<<<MGROUPS * 12, 128, 0, stream>>>(aoh, wprojh, proj_b, (float*)d_out);
}
